// BLSTM_31945966747897
// MI455X (gfx1250) — matrix operand order repaired and checked by reading, not compiled, not run on hardware
//
#include <hip/hip_runtime.h>
#include <math.h>

// ---------------- CDNA5 WMMA types ----------------
typedef __attribute__((ext_vector_type(16))) __bf16       v16bf;
typedef __attribute__((ext_vector_type(8)))  float        v8f;
typedef __attribute__((ext_vector_type(4)))  unsigned int u32x4;

// ---------------- problem constants ----------------
#define U_DIM   512
#define GDIM    2048          // 4*U
#define BB      4
#define TT      4200
#define WIDTH_  200
#define STRIDE_ 100
#define FJ      41            // frames per batch
#define NSEQ    164           // BB*FJ
#define NSEQ_PAD 176          // padded to multiple of 16
#define NROWS   32800         // NSEQ*WIDTH_ (multiple of 16)

__device__ __forceinline__ unsigned short f2bf(float f) {
  unsigned u = __builtin_bit_cast(unsigned, f);
  u += 0x7FFFu + ((u >> 16) & 1u);           // round-to-nearest-even
  return (unsigned short)(u >> 16);
}
__device__ __forceinline__ __bf16 f2bf16(float f) {
  unsigned short s = f2bf(f);
  return __builtin_bit_cast(__bf16, s);
}

__device__ __forceinline__ v16bf load_frag16(const __bf16* p0, const __bf16* p1) {
  union { u32x4 u[2]; v16bf v; } t;
  t.u[0] = *(const u32x4*)p0;
  t.u[1] = *(const u32x4*)p1;
  return t.v;
}

// ---------------- frame extraction: (B,U,T) f32 -> (NROWS,512) bf16 ----------------
__global__ void blstm_frame_pack(const float* __restrict__ in, __bf16* __restrict__ X0) {
  int idx = blockIdx.x * blockDim.x + threadIdx.x;
  if (idx >= NROWS * U_DIM) return;
  int u = idx & (U_DIM - 1);
  int r = idx >> 9;                 // frame row = seq*WIDTH + w
  int w = r % WIDTH_;
  int s = r / WIDTH_;
  int f = s % FJ;
  int b = s / FJ;
  float v = in[(size_t)(b * U_DIM + u) * TT + (size_t)(f * STRIDE_ + w)];
  X0[idx] = f2bf16(v);
}

// ---------------- pack f32 KxN row-major weight into wave32 B-fragment layout ----------------
// fragment (kt,nt): 32 lanes x 16 bf16 contiguous. lane L: col = nt*16+(L&15),
// K = kt*32 + (L<16?0:8) + (j<8 ? j : j+8), j=0..15  (matches V_WMMA bf16 B layout)
__global__ void blstm_pack_b(const float* __restrict__ W, __bf16* __restrict__ out, int K, int N) {
  int idx = blockIdx.x * blockDim.x + threadIdx.x;
  if (idx >= K * N) return;
  int j    = idx & 15;
  int lane = (idx >> 4) & 31;
  int frag = idx >> 9;              // kt*Ntiles + ntile
  int Ntiles = N >> 4;
  int ntile = frag % Ntiles;
  int kt    = frag / Ntiles;
  int n = ntile * 16 + (lane & 15);
  int k = kt * 32 + ((lane < 16) ? 0 : 8) + ((j < 8) ? j : j + 8);
  out[idx] = f2bf16(W[(size_t)k * N + n]);
}

// ---------------- bf16 WMMA GEMM: C = A@Bp (+bias) (+addend) ----------------
// One wave computes a 16x64 tile (1 M-tile x 4 N-tiles): A-frag reuse x4,
// 4 independent WMMA accumulator chains, double-buffered fragment pipeline.
// 8 waves / block. A row-major bf16 (lda even), Bp packed fragments.
__global__ void blstm_wmma_gemm(const __bf16* __restrict__ A, int lda,
                                const __bf16* __restrict__ Bp,
                                const float* __restrict__ bias,
                                const float* __restrict__ addend,
                                long long addStride, int addMreal,
                                float* __restrict__ C, int ldc,
                                int Mtiles, int Ngroups, int K) {
  int wave = blockIdx.x * 8 + (threadIdx.x >> 5);
  if (wave >= Mtiles * Ngroups) return;         // wave-uniform: EXEC stays all-ones
  int ng = wave % Ngroups;                      // group of 4 consecutive N tiles
  int mt = wave / Ngroups;
  int lane = threadIdx.x & 31;
  int halfsel = lane >> 4;                      // lanes 0-15 vs 16-31
  int l16 = lane & 15;
  int Ntiles = Ngroups * 4;

  // A-fragment base: lane holds row (mt*16+l16); K runs {h*8..+7} and {h*8+16..+23}
  const __bf16* arow  = A + (size_t)(mt * 16 + l16) * lda + halfsel * 8;
  // B fragments for the 4 N-tiles are 4KB contiguous per kt in packed layout
  const __bf16* bbase = Bp + (((size_t)ng * 4) << 9) + lane * 16;
  size_t bkt = (size_t)Ntiles << 9;             // packed elements per kt step

  v8f acc0 = {}, acc1 = {}, acc2 = {}, acc3 = {};
  int nkt = K >> 5;

  // prologue: fragments for kt = 0
  v16bf a_c  = load_frag16(arow, arow + 16);
  v16bf b_c0 = load_frag16(bbase,                bbase + 8);
  v16bf b_c1 = load_frag16(bbase + 512,          bbase + 512 + 8);
  v16bf b_c2 = load_frag16(bbase + 1024,         bbase + 1024 + 8);
  v16bf b_c3 = load_frag16(bbase + 1536,         bbase + 1536 + 8);

  for (int kt = 1; kt < nkt; ++kt) {
    // issue next-iteration loads before consuming current fragments
    const __bf16* ap = arow + kt * 32;
    const __bf16* bp = bbase + (size_t)kt * bkt;
    v16bf a_n  = load_frag16(ap, ap + 16);
    v16bf b_n0 = load_frag16(bp,        bp + 8);
    v16bf b_n1 = load_frag16(bp + 512,  bp + 512 + 8);
    v16bf b_n2 = load_frag16(bp + 1024, bp + 1024 + 8);
    v16bf b_n3 = load_frag16(bp + 1536, bp + 1536 + 8);

    acc0 = __builtin_amdgcn_wmma_f32_16x16x32_bf16(false, a_c, false, b_c0, (short)0, acc0, false, false);
    acc1 = __builtin_amdgcn_wmma_f32_16x16x32_bf16(false, a_c, false, b_c1, (short)0, acc1, false, false);
    acc2 = __builtin_amdgcn_wmma_f32_16x16x32_bf16(false, a_c, false, b_c2, (short)0, acc2, false, false);
    acc3 = __builtin_amdgcn_wmma_f32_16x16x32_bf16(false, a_c, false, b_c3, (short)0, acc3, false, false);

    a_c = a_n; b_c0 = b_n0; b_c1 = b_n1; b_c2 = b_n2; b_c3 = b_n3;
  }
  acc0 = __builtin_amdgcn_wmma_f32_16x16x32_bf16(false, a_c, false, b_c0, (short)0, acc0, false, false);
  acc1 = __builtin_amdgcn_wmma_f32_16x16x32_bf16(false, a_c, false, b_c1, (short)0, acc1, false, false);
  acc2 = __builtin_amdgcn_wmma_f32_16x16x32_bf16(false, a_c, false, b_c2, (short)0, acc2, false, false);
  acc3 = __builtin_amdgcn_wmma_f32_16x16x32_bf16(false, a_c, false, b_c3, (short)0, acc3, false, false);

  // epilogue: C layout per tile: VGPR v -> M = v (lanes 0-15) / 8+v (lanes 16-31), N = l16
  v8f accs[4] = {acc0, acc1, acc2, acc3};
  #pragma unroll
  for (int i = 0; i < 4; ++i) {
    int n = (ng * 4 + i) * 16 + l16;
    float bv = bias ? bias[n] : 0.0f;
    #pragma unroll
    for (int v = 0; v < 8; ++v) {
      int m = mt * 16 + halfsel * 8 + v;
      float r = accs[i][v] + bv;
      if (addend && m < addMreal) r += addend[(size_t)m * addStride + n];
      C[(size_t)m * ldc + n] = r;
    }
  }
}

// ---------------- LSTM gate pointwise ----------------
__global__ void blstm_gate(const float* __restrict__ z,       // NSEQ_PAD x 2048
                           float* __restrict__ c,             // NSEQ x 512
                           __bf16* __restrict__ h,            // NSEQ_PAD x 512 (rows>=NSEQ stay 0)
                           __bf16* __restrict__ Hout, int ldh, int coloff, int t) {
  int idx = blockIdx.x * blockDim.x + threadIdx.x;
  if (idx >= NSEQ * U_DIM) return;
  int j = idx & (U_DIM - 1);
  int n = idx >> 9;
  const float* zr = z + (size_t)n * GDIM;
  float ig = 1.0f / (1.0f + __expf(-zr[j]));
  float fg = 1.0f / (1.0f + __expf(-zr[U_DIM + j]));
  float gg = tanhf(zr[2 * U_DIM + j]);
  float og = 1.0f / (1.0f + __expf(-zr[3 * U_DIM + j]));
  float cc = fg * c[idx] + ig * gg;
  c[idx] = cc;
  __bf16 hb = f2bf16(og * tanhf(cc));
  h[(size_t)n * U_DIM + j] = hb;
  Hout[(size_t)(n * WIDTH_ + t) * ldh + coloff + j] = hb;
}

// ---------------- scan state init ----------------
__global__ void blstm_scan_init(__bf16* __restrict__ h, float* __restrict__ c) {
  int idx = blockIdx.x * blockDim.x + threadIdx.x;
  if (idx < NSEQ_PAD * U_DIM) h[idx] = __builtin_bit_cast(__bf16, (unsigned short)0);
  if (idx < NSEQ * U_DIM)     c[idx] = 0.0f;
}

// ---------------- overlap-add + residual: out[b][u][t] ----------------
__global__ void blstm_overlap_add(const float* __restrict__ in, const float* __restrict__ P,
                                  float* __restrict__ out) {
  int idx = blockIdx.x * blockDim.x + threadIdx.x;
  if (idx >= BB * U_DIM * TT) return;
  int t = idx % TT;
  int r = idx / TT;
  int u = r & (U_DIM - 1);
  int b = r >> 9;
  float s = in[idx];
  int fhi = t / STRIDE_; if (fhi > FJ - 1) fhi = FJ - 1;
  int flo = (t >= WIDTH_) ? ((t - WIDTH_) / STRIDE_ + 1) : 0;
  for (int f = flo; f <= fhi; ++f) {
    int w = t - f * STRIDE_;
    s += P[(size_t)((b * FJ + f) * WIDTH_ + w) * U_DIM + u];
  }
  out[idx] = s;
}

// ---------------- host orchestration ----------------
extern "C" void kernel_launch(void* const* d_in, const int* in_sizes, int n_in,
                              void* d_out, int out_size, void* d_ws, size_t ws_size,
                              hipStream_t stream) {
  (void)in_sizes; (void)n_in; (void)out_size; (void)ws_size;
  const float* in = (const float*)d_in[0];
  const float* Wp = (const float*)d_in[13];
  const float* bp = (const float*)d_in[14];

  char* w = (char*)d_ws;
  auto carve = [&](size_t bytes) { char* p = w; w += (bytes + 255) & ~(size_t)255; return p; };

  __bf16* X0  = (__bf16*)carve((size_t)NROWS * U_DIM * 2);       // frames bf16
  __bf16* H0  = (__bf16*)carve((size_t)NROWS * U_DIM * 2);       // layer0 output (per dir)
  __bf16* HC  = (__bf16*)carve((size_t)NROWS * 2 * U_DIM * 2);   // concat fwd|bwd
  float*  Z   = (float*) carve((size_t)NROWS * GDIM * 4);        // zx (reused per layer-dir)
  float*  P   = (float*) carve((size_t)NROWS * U_DIM * 4);       // projection
  __bf16* Wxp[4]; __bf16* Whp[4];
  for (int i = 0; i < 4; ++i) Wxp[i] = (__bf16*)carve((size_t)U_DIM * GDIM * 2);
  for (int i = 0; i < 4; ++i) Whp[i] = (__bf16*)carve((size_t)U_DIM * GDIM * 2);
  __bf16* Wpp  = (__bf16*)carve((size_t)(2 * U_DIM) * U_DIM * 2);
  float*  zbuf = (float*) carve((size_t)NSEQ_PAD * GDIM * 4);
  __bf16* hbuf = (__bf16*)carve((size_t)NSEQ_PAD * U_DIM * 2);
  float*  cbuf = (float*) carve((size_t)NSEQ * U_DIM * 4);

  const int TB = 256;

  // frames
  blstm_frame_pack<<<(NROWS * U_DIM + TB - 1) / TB, TB, 0, stream>>>(in, X0);

  // pack all weights into B-fragment layout
  for (int d = 0; d < 2; ++d)
    for (int l = 0; l < 2; ++l) {
      int base = 1 + d * 6 + l * 3;
      blstm_pack_b<<<(U_DIM * GDIM + TB - 1) / TB, TB, 0, stream>>>(
          (const float*)d_in[base + 0], Wxp[d * 2 + l], U_DIM, GDIM);
      blstm_pack_b<<<(U_DIM * GDIM + TB - 1) / TB, TB, 0, stream>>>(
          (const float*)d_in[base + 1], Whp[d * 2 + l], U_DIM, GDIM);
    }
  blstm_pack_b<<<(2 * U_DIM * U_DIM + TB - 1) / TB, TB, 0, stream>>>(Wp, Wpp, 2 * U_DIM, U_DIM);

  // LSTM stacks (fwd d=0, bwd d=1)
  for (int d = 0; d < 2; ++d) {
    for (int l = 0; l < 2; ++l) {
      int base = 1 + d * 6 + l * 3;
      const float* bias = (const float*)d_in[base + 2];
      const __bf16* Ain = (l == 0) ? X0 : H0;
      __bf16* Hout = (l == 0) ? H0 : HC;
      int ldh = (l == 0) ? U_DIM : 2 * U_DIM;
      int coloff = (l == 0) ? 0 : d * U_DIM;

      // zx = Ain @ Wx + b : M=32800 (2050 tiles), N=2048 (32 groups), K=512
      blstm_wmma_gemm<<<(2050 * 32) / 8, TB, 0, stream>>>(
          Ain, U_DIM, Wxp[d * 2 + l], bias, nullptr, 0, 0, Z, GDIM, 2050, 32, U_DIM);

      blstm_scan_init<<<(NSEQ_PAD * U_DIM + TB - 1) / TB, TB, 0, stream>>>(hbuf, cbuf);

      for (int step = 0; step < WIDTH_; ++step) {
        int t = (d == 0) ? step : (WIDTH_ - 1 - step);
        // z_t = h @ Wh + zx[:, t, :] : M=176 (11 tiles), N=2048 (32 groups), K=512
        blstm_wmma_gemm<<<(11 * 32) / 8, TB, 0, stream>>>(
            hbuf, U_DIM, Whp[d * 2 + l], nullptr,
            Z + (size_t)t * GDIM, (long long)WIDTH_ * GDIM, NSEQ,
            zbuf, GDIM, 11, 32, U_DIM);
        blstm_gate<<<(NSEQ * U_DIM + TB - 1) / TB, TB, 0, stream>>>(
            zbuf, cbuf, hbuf, Hout, ldh, coloff, t);
      }
    }
  }

  // projection: P = HC @ Wp + bp : M=32800 (2050 tiles), N=512 (8 groups), K=1024
  blstm_wmma_gemm<<<(2050 * 8) / 8, TB, 0, stream>>>(
      HC, 2 * U_DIM, Wpp, bp, nullptr, 0, 0, P, U_DIM, 2050, 8, 2 * U_DIM);

  // overlap-add + residual
  blstm_overlap_add<<<(BB * U_DIM * TT + TB - 1) / TB, TB, 0, stream>>>(in, P, (float*)d_out);
}
